// EquiformerV2Block_33036888441071
// MI455X (gfx1250) — matrix operand order repaired and checked by reading, not compiled, not run on hardware
//
#include <hip/hip_runtime.h>
#include <hip/hip_bf16.h>

// CDNA5 / gfx1250, wave32. bf16 WMMA GEMM pipeline for EquiformerV2 block.

typedef __attribute__((ext_vector_type(16))) __bf16 v16bf;
typedef __attribute__((ext_vector_type(8)))  float  v8f;
typedef __attribute__((ext_vector_type(4)))  unsigned short us4;

#define NN    4096
#define KNNB  16
#define CCH   128
#define LLDIM 9
#define EE    (NN*KNNB)
#define NPBN  1024
#define CUTF  3.0f

__device__ inline float siluf_(float x){ return x / (1.0f + __expf(-x)); }
__device__ inline unsigned short f2bf(float f){
    union { float f; unsigned u; } v; v.f = f;
    unsigned r = v.u + 0x7fffu + ((v.u >> 16) & 1u);
    return (unsigned short)(r >> 16);
}
__device__ inline us4 f2bf4(float4 f){
    us4 o; o.x = f2bf(f.x); o.y = f2bf(f.y); o.z = f2bf(f.z); o.w = f2bf(f.w); return o;
}
__device__ inline __bf16 bfbits(unsigned short h){ __bf16 b; __builtin_memcpy(&b, &h, 2); return b; }

// ---------------------------------------------------------------- kNN (top-16 per node, per 1024-node batch)
__global__ __launch_bounds__(128) void knn_kernel(const float* __restrict__ coord,
                                                  int* __restrict__ esrc, float* __restrict__ elen){
    __shared__ float d2s[NPBN];
    __shared__ unsigned long long red[128];
    int i = blockIdx.x, t = threadIdx.x;
    int base = (i >> 10) << 10;
    float cx = coord[i*3+0], cy = coord[i*3+1], cz = coord[i*3+2];
    for (int j = t; j < NPBN; j += 128){
        int g = base + j;
        float dx = cx - coord[g*3+0], dy = cy - coord[g*3+1], dz = cz - coord[g*3+2];
        float d2 = dx*dx + dy*dy + dz*dz;
        if (g == i) d2 = 1e9f;
        d2s[j] = d2;
    }
    __syncthreads();
    for (int r = 0; r < KNNB; ++r){
        unsigned long long best = ~0ull;
        for (int j = t; j < NPBN; j += 128){
            union { float f; unsigned u; } v; v.f = d2s[j];
            unsigned long long key = ((unsigned long long)v.u << 32) | (unsigned)j;
            if (key < best) best = key;
        }
        red[t] = best;
        __syncthreads();
        for (int s = 64; s > 0; s >>= 1){
            if (t < s && red[t+s] < red[t]) red[t] = red[t+s];
            __syncthreads();
        }
        if (t == 0){
            unsigned long long k = red[0];
            int jm = (int)(k & 0xffffffffu);
            union { unsigned u; float f; } v; v.u = (unsigned)(k >> 32);
            esrc[i*KNNB + r] = base + jm;
            elen[i*KNNB + r] = sqrtf(fmaxf(v.f, 0.0f));
            d2s[jm] = 1e9f;
        }
        __syncthreads();
    }
}

// ---------------------------------------------------------------- generic bf16 WMMA GEMM
// C[M,N] = epilogue(A[M,K](lda) @ B[K,N]); tile 128x64, 256 thr = 8 waves, one 16-row strip/wave.
// Epilogue statically selected: RESID adds resid[gr*N+gc]; GATE multiplies gate[(gr/gdiv)*N+gc];
// ACT==1 applies SiLU before gate/resid.
template<bool RESID, bool GATE, int ACT>
__global__ __launch_bounds__(256) void gemm_bf16_kernel(
        const float* __restrict__ A, int lda, const float* __restrict__ B,
        const float* __restrict__ resid, const float* __restrict__ gate, int gdiv,
        float* __restrict__ C, int M, int N, int K)
{
    __shared__ unsigned short As[128][36];
    __shared__ unsigned short Bs[64][36];
    int tid = threadIdx.x, wave = tid >> 5, lane = tid & 31;
    int rowBase = blockIdx.x * 128, colBase = blockIdx.y * 64;
    v8f acc[4];
    { v8f z = {}; for (int q = 0; q < 4; ++q) acc[q] = z; }

    for (int k0 = 0; k0 < K; k0 += 32){
        // A tile: 128x32 floats = 1024 float4 loads -> bf16x4 LDS stores (8B aligned)
        #pragma unroll
        for (int it = 0; it < 4; ++it){
            int idx = tid + it*256;
            int r = idx >> 3, c4 = (idx & 7) << 2;
            float4 av = *(const float4*)(A + (size_t)(rowBase + r) * lda + k0 + c4);
            *(us4*)&As[r][c4] = f2bf4(av);
        }
        // B tile: 32x64 floats = 512 float4 loads -> transposed scalar LDS stores
        #pragma unroll
        for (int it = 0; it < 2; ++it){
            int idx = tid + it*256;
            int kk = idx >> 4, c4 = (idx & 15) << 2;
            float4 bv = *(const float4*)(B + (size_t)(k0 + kk) * N + colBase + c4);
            Bs[c4+0][kk] = f2bf(bv.x);
            Bs[c4+1][kk] = f2bf(bv.y);
            Bs[c4+2][kk] = f2bf(bv.z);
            Bs[c4+3][kk] = f2bf(bv.w);
        }
        __syncthreads();
        int arow = wave*16 + (lane & 15);
        int akb  = (lane >> 4) << 4;
        v16bf af;
        #pragma unroll
        for (int j = 0; j < 8; ++j){
            af[j]   = bfbits(As[arow][akb + j]);
            af[j+8] = bfbits(As[arow][akb + 8 + j]);
        }
        #pragma unroll
        for (int tc = 0; tc < 4; ++tc){
            int bcol = tc*16 + (lane & 15);
            int bkb  = (lane >> 4) << 4;
            v16bf bf;
            #pragma unroll
            for (int j = 0; j < 16; ++j) bf[j] = bfbits(Bs[bcol][bkb + j]);
            acc[tc] = __builtin_amdgcn_wmma_f32_16x16x32_bf16(
                false, af, false, bf, (short)0, acc[tc], false, false);
        }
        __syncthreads();
    }
    int rit = (lane >> 4) << 3;
    #pragma unroll
    for (int tc = 0; tc < 4; ++tc){
        int gc = colBase + tc*16 + (lane & 15);
        #pragma unroll
        for (int j = 0; j < 8; ++j){
            int gr = rowBase + wave*16 + rit + j;
            float v = acc[tc][j];
            if (ACT == 1) v = siluf_(v);
            if (GATE)  v *= gate[(size_t)(gr / gdiv) * N + gc];
            if (RESID) v += resid[(size_t)gr * N + gc];
            C[(size_t)gr * N + gc] = v;
        }
    }
}

// ---------------------------------------------------------------- edge MLP: edge_h = silu(rbf@W1+b1)@W2+b2
// RBF computed on the fly into LDS bf16; two chained WMMA stages; 128 edges/block.
__global__ __launch_bounds__(256) void edge_mlp_kernel(
        const float* __restrict__ elen,
        const float* __restrict__ W1, const float* __restrict__ b1,
        const float* __restrict__ W2, const float* __restrict__ b2,
        float* __restrict__ eh)
{
    __shared__ unsigned short Rs[128][132];
    __shared__ unsigned short Bs[128][36];
    int tid = threadIdx.x, wave = tid >> 5, lane = tid & 31;
    int e0 = blockIdx.x * 128;
    for (int idx = tid; idx < 128*128; idx += 256){
        int r = idx >> 7, c = idx & 127;
        float len = elen[e0 + r];
        float d = (len - (CUTF/127.0f) * (float)c) * (128.0f / CUTF);
        Rs[r][c] = f2bf(__expf(-0.5f * d * d));
    }
    __syncthreads();

    v8f acc[8];
    { v8f z = {}; for (int q = 0; q < 8; ++q) acc[q] = z; }
    int arow = wave*16 + (lane & 15);
    int rit  = (lane >> 4) << 3;

    // stage 1: hidden = rbf @ W1
    for (int k0 = 0; k0 < 128; k0 += 32){
        #pragma unroll
        for (int it = 0; it < 4; ++it){
            int idx = tid + it*256;
            int kk = idx >> 5, c4 = (idx & 31) << 2;
            float4 wv = *(const float4*)(W1 + (size_t)(k0 + kk)*128 + c4);
            Bs[c4+0][kk] = f2bf(wv.x);
            Bs[c4+1][kk] = f2bf(wv.y);
            Bs[c4+2][kk] = f2bf(wv.z);
            Bs[c4+3][kk] = f2bf(wv.w);
        }
        __syncthreads();
        int akb = k0 + ((lane >> 4) << 4);
        v16bf af;
        #pragma unroll
        for (int j = 0; j < 8; ++j){
            af[j]   = bfbits(Rs[arow][akb + j]);
            af[j+8] = bfbits(Rs[arow][akb + 8 + j]);
        }
        #pragma unroll
        for (int tc = 0; tc < 8; ++tc){
            int bcol = tc*16 + (lane & 15);
            int bkb  = (lane >> 4) << 4;
            v16bf bf;
            #pragma unroll
            for (int j = 0; j < 16; ++j) bf[j] = bfbits(Bs[bcol][bkb + j]);
            acc[tc] = __builtin_amdgcn_wmma_f32_16x16x32_bf16(
                false, af, false, bf, (short)0, acc[tc], false, false);
        }
        __syncthreads();
    }
    // silu(acc + b1) -> Rs (reuse as hidden activations, bf16)
    #pragma unroll
    for (int tc = 0; tc < 8; ++tc){
        int col = tc*16 + (lane & 15);
        float bb = b1[col];
        #pragma unroll
        for (int j = 0; j < 8; ++j){
            int row = wave*16 + rit + j;
            Rs[row][col] = f2bf(siluf_(acc[tc][j] + bb));
        }
    }
    __syncthreads();

    { v8f z = {}; for (int q = 0; q < 8; ++q) acc[q] = z; }
    // stage 2: edge_h = hidden @ W2
    for (int k0 = 0; k0 < 128; k0 += 32){
        #pragma unroll
        for (int it = 0; it < 4; ++it){
            int idx = tid + it*256;
            int kk = idx >> 5, c4 = (idx & 31) << 2;
            float4 wv = *(const float4*)(W2 + (size_t)(k0 + kk)*128 + c4);
            Bs[c4+0][kk] = f2bf(wv.x);
            Bs[c4+1][kk] = f2bf(wv.y);
            Bs[c4+2][kk] = f2bf(wv.z);
            Bs[c4+3][kk] = f2bf(wv.w);
        }
        __syncthreads();
        int akb = k0 + ((lane >> 4) << 4);
        v16bf af;
        #pragma unroll
        for (int j = 0; j < 8; ++j){
            af[j]   = bfbits(Rs[arow][akb + j]);
            af[j+8] = bfbits(Rs[arow][akb + 8 + j]);
        }
        #pragma unroll
        for (int tc = 0; tc < 8; ++tc){
            int bcol = tc*16 + (lane & 15);
            int bkb  = (lane >> 4) << 4;
            v16bf bf;
            #pragma unroll
            for (int j = 0; j < 16; ++j) bf[j] = bfbits(Bs[bcol][bkb + j]);
            acc[tc] = __builtin_amdgcn_wmma_f32_16x16x32_bf16(
                false, af, false, bf, (short)0, acc[tc], false, false);
        }
        __syncthreads();
    }
    #pragma unroll
    for (int tc = 0; tc < 8; ++tc){
        int col = tc*16 + (lane & 15);
        float bb = b2[col];
        #pragma unroll
        for (int j = 0; j < 8; ++j){
            int row = wave*16 + rit + j;
            eh[((size_t)(e0 + row))*128 + col] = acc[tc][j] + bb;
        }
    }
}

// ---------------------------------------------------------------- attention + aggregation (rotations cancel)
__global__ __launch_bounds__(128) void attn_agg_kernel(
        const int* __restrict__ esrc, const float* __restrict__ elen,
        const float* __restrict__ qs, const float* __restrict__ qd,
        const float* __restrict__ wa, const float* __restrict__ eh,
        const float* __restrict__ y, float* __restrict__ agg)
{
    __shared__ int   src_s[16];
    __shared__ float len_s[16];
    __shared__ float logit_s[16][8];
    int n = blockIdx.x, t = threadIdx.x;
    if (t < 16){ src_s[t] = esrc[n*16 + t]; len_s[t] = elen[n*16 + t]; }
    __syncthreads();
    {   // 128 threads = 16 edges x 8 heads
        int k = t >> 3, h = t & 7;
        int sk = src_s[k];
        float acc = 0.0f;
        #pragma unroll
        for (int a = 0; a < 8; ++a){
            float qv = qd[n*64 + h*8 + a] + qs[sk*64 + h*8 + a];
            acc += siluf_(qv) * wa[h*8 + a];
        }
        if (len_s[k] > CUTF) acc = -1e9f;
        logit_s[k][h] = acc;
    }
    __syncthreads();
    if (t < 8){
        int h = t;
        float m = -1e30f;
        #pragma unroll
        for (int k = 0; k < 16; ++k) m = fmaxf(m, logit_s[k][h]);
        float ex[16], den = 0.0f;
        #pragma unroll
        for (int k = 0; k < 16; ++k){ ex[k] = __expf(logit_s[k][h] - m); den += ex[k]; }
        float inv = 1.0f / (den + 1e-9f);
        #pragma unroll
        for (int k = 0; k < 16; ++k) logit_s[k][h] = ex[k] * inv;
    }
    __syncthreads();
    int c = t, h = c >> 4;           // head = channel / V (V=16)
    float w[16];
    #pragma unroll
    for (int k = 0; k < 16; ++k)
        w[k] = logit_s[k][h] * eh[((size_t)n*16 + k)*128 + c];
    #pragma unroll
    for (int l = 0; l < LLDIM; ++l){
        float a = 0.0f;
        #pragma unroll
        for (int k = 0; k < 16; ++k)
            a += w[k] * y[((size_t)src_s[k]*LLDIM + l)*128 + c];
        agg[((size_t)n*LLDIM + l)*128 + c] = a;
    }
}

// ---------------------------------------------------------------- RMS norm over 128-channel rows
__global__ __launch_bounds__(128) void rmsnorm_kernel(const float* __restrict__ x,
                                                      const float* __restrict__ gamma,
                                                      float* __restrict__ out){
    __shared__ float red[128];
    int r = blockIdx.x, t = threadIdx.x;
    float v = x[(size_t)r*128 + t];
    red[t] = v * v;
    __syncthreads();
    for (int s = 64; s > 0; s >>= 1){ if (t < s) red[t] += red[t+s]; __syncthreads(); }
    float sc = sqrtf(red[0] * (1.0f/128.0f) + 1e-6f);
    out[(size_t)r*128 + t] = v / sc * gamma[t];
}

// ---------------------------------------------------------------- FiLM coefficients: mish(cond) @ W_film + b
__global__ __launch_bounds__(256) void film_kernel(const float* __restrict__ cond,
                                                   const float* __restrict__ Wf,
                                                   const float* __restrict__ bf_,
                                                   float* __restrict__ film){
    __shared__ float mc[256];
    int b = blockIdx.x, t = threadIdx.x;
    float x = cond[b*256 + t];
    float sp = __logf(1.0f + __expf(x));
    mc[t] = x * tanhf(sp);
    __syncthreads();
    for (int j = t; j < 2304; j += 256){
        float s = bf_[j];
        for (int k = 0; k < 256; ++k) s += mc[k] * Wf[k*2304 + j];
        film[b*2304 + j] = s;
    }
}

// ---------------------------------------------------------------- final RMS norm + FiLM, write output block
__global__ __launch_bounds__(128) void finalize_kernel(const float* __restrict__ x,
                                                       const float* __restrict__ gamma1,
                                                       const float* __restrict__ film,
                                                       float* __restrict__ out){
    __shared__ float red[128];
    int r = blockIdx.x, t = threadIdx.x;
    int n = r / LLDIM, l = r - n*LLDIM, b = n >> 10;
    float v = x[(size_t)r*128 + t];
    red[t] = v * v;
    __syncthreads();
    for (int s = 64; s > 0; s >>= 1){ if (t < s) red[t] += red[t+s]; __syncthreads(); }
    float sc = sqrtf(red[0] * (1.0f/128.0f) + 1e-6f);
    float xn = v / sc * gamma1[t];
    int j = l*128 + t;
    out[12288 + (size_t)r*128 + t] = xn * film[b*2304 + j] + film[b*2304 + 1152 + j];
}

// ---------------------------------------------------------------- copy coord + batch into output tuple
__global__ __launch_bounds__(256) void copy_misc_kernel(const float* __restrict__ coord,
                                                        const int* __restrict__ batch,
                                                        float* __restrict__ out){
    int i = blockIdx.x * 256 + threadIdx.x;
    if (i < 12288) out[i] = coord[i];
    if (i < 4096)  ((int*)out)[12288 + 4718592 + i] = batch[i];
}

// ================================================================ launch
extern "C" void kernel_launch(void* const* d_in, const int* in_sizes, int n_in,
                              void* d_out, int out_size, void* d_ws, size_t ws_size,
                              hipStream_t stream)
{
    const float* coord  = (const float*)d_in[0];
    const float* emb    = (const float*)d_in[1];
    const int*   batch  = (const int*)  d_in[2];
    const float* cond   = (const float*)d_in[3];
    const float* W_e1   = (const float*)d_in[4];
    const float* b_e1   = (const float*)d_in[5];
    const float* W_e2   = (const float*)d_in[6];
    const float* b_e2   = (const float*)d_in[7];
    const float* W_src  = (const float*)d_in[8];
    const float* W_dst  = (const float*)d_in[9];
    const float* w_a    = (const float*)d_in[10];
    const float* W_v    = (const float*)d_in[11];
    const float* W_o    = (const float*)d_in[12];
    const float* g1     = (const float*)d_in[13];
    const float* g2     = (const float*)d_in[14];
    const float* W_g    = (const float*)d_in[15];
    const float* W_f1   = (const float*)d_in[16];
    const float* W_f2   = (const float*)d_in[17];
    const float* W_film = (const float*)d_in[18];
    const float* b_film = (const float*)d_in[19];

    char* ws = (char*)d_ws;
    size_t off = 0;
    auto alloc = [&](size_t bytes){ size_t o = off; off += (bytes + 255) & ~(size_t)255; return o; };
    int*   esrc = (int*)  (ws + alloc((size_t)EE * 4));
    float* elen = (float*)(ws + alloc((size_t)EE * 4));
    float* qs   = (float*)(ws + alloc((size_t)NN * 64 * 4));
    float* qd   = (float*)(ws + alloc((size_t)NN * 64 * 4));
    float* y    = (float*)(ws + alloc((size_t)NN * LLDIM * 128 * 4));
    float* eh   = (float*)(ws + alloc((size_t)EE * 128 * 4));
    float* agg  = (float*)(ws + alloc((size_t)NN * LLDIM * 128 * 4));
    float* x1   = (float*)(ws + alloc((size_t)NN * LLDIM * 128 * 4));
    float* hbuf = (float*)(ws + alloc((size_t)NN * LLDIM * 128 * 4));
    float* gate = (float*)(ws + alloc((size_t)NN * 256 * 4));
    float* hid  = (float*)(ws + alloc((size_t)NN * LLDIM * 256 * 4));
    float* film = (float*)(ws + alloc((size_t)4 * 2304 * 4));
    (void)in_sizes; (void)n_in; (void)out_size; (void)ws_size;

    const int MR = NN * LLDIM;   // 36864 rows (node,l)

    knn_kernel<<<NN, 128, 0, stream>>>(coord, esrc, elen);

    // y = emb @ W_v (rotation/W_v commute with per-edge scaling)
    gemm_bf16_kernel<false,false,0><<<dim3(MR/128, 2), 256, 0, stream>>>(
        emb, 128, W_v, nullptr, nullptr, 1, y, MR, 128, 128);
    // q_src = emb[:,0] @ W_src ; q_dst = emb[:,0] @ W_dst  (lda = L*C)
    gemm_bf16_kernel<false,false,0><<<dim3(NN/128, 1), 256, 0, stream>>>(
        emb, 1152, W_src, nullptr, nullptr, 1, qs, NN, 64, 128);
    gemm_bf16_kernel<false,false,0><<<dim3(NN/128, 1), 256, 0, stream>>>(
        emb, 1152, W_dst, nullptr, nullptr, 1, qd, NN, 64, 128);

    edge_mlp_kernel<<<EE/128, 256, 0, stream>>>(elen, W_e1, b_e1, W_e2, b_e2, eh);

    attn_agg_kernel<<<NN, 128, 0, stream>>>(esrc, elen, qs, qd, w_a, eh, y, agg);

    // x1 = emb + agg @ W_o
    gemm_bf16_kernel<true,false,0><<<dim3(MR/128, 2), 256, 0, stream>>>(
        agg, 128, W_o, emb, nullptr, 1, x1, MR, 128, 128);
    // h = rmsnorm(x1, gamma2)
    rmsnorm_kernel<<<MR, 128, 0, stream>>>(x1, g2, hbuf);
    // gate = silu(h[:,0] @ W_g)
    gemm_bf16_kernel<false,false,1><<<dim3(NN/128, 4), 256, 0, stream>>>(
        hbuf, 1152, W_g, nullptr, nullptr, 1, gate, NN, 256, 128);
    // hid = (h @ W_f1) * gate[n]
    gemm_bf16_kernel<false,true,0><<<dim3(MR/128, 4), 256, 0, stream>>>(
        hbuf, 128, W_f1, nullptr, gate, LLDIM, hid, MR, 256, 128);
    // x1 = x1 + hid @ W_f2   (out aliases residual; each element read then written once)
    gemm_bf16_kernel<true,false,0><<<dim3(MR/128, 2), 256, 0, stream>>>(
        hid, 256, W_f2, x1, nullptr, 1, x1, MR, 128, 256);

    film_kernel<<<4, 256, 0, stream>>>(cond, W_film, b_film, film);
    finalize_kernel<<<MR, 128, 0, stream>>>(x1, g1, film, (float*)d_out);
    copy_misc_kernel<<<64, 256, 0, stream>>>(coord, batch, (float*)d_out);
}
